// MultiHeadSelfAttention_38611755991241
// MI455X (gfx1250) — matrix operand order repaired and checked by reading, not compiled, not run on hardware
//
#include <hip/hip_runtime.h>

#define D_MODEL   1024
#define NUM_HEADS 16
#define HEAD_DIM  64
#define BATCH     4
#define SEQ       2048
#define M_ROWS    (BATCH * SEQ)   // 8192

typedef unsigned short u16;
typedef __attribute__((ext_vector_type(16))) __bf16 v16bf;
typedef __attribute__((ext_vector_type(8)))  float  v8f;
typedef __attribute__((ext_vector_type(4)))  int    v4i;

union Frag16 {
  v16bf bf;
  u16   u[16];
  uint4 q[2];
};

__device__ __forceinline__ u16 f32_to_bf16(float f) {
  unsigned int x = __float_as_uint(f);
  x += 0x7FFFu + ((x >> 16) & 1u);   // round-to-nearest-even
  return (u16)(x >> 16);
}

__device__ __forceinline__ v8f zero8() {
  v8f z;
  #pragma unroll
  for (int i = 0; i < 8; ++i) z[i] = 0.0f;
  return z;
}

// ---- gfx1250 async global->LDS staging (ASYNCcnt path), with fallback ------
#if defined(__gfx1250__) && __has_builtin(__builtin_amdgcn_global_load_async_to_lds_b128)
#define HAVE_ASYNC_LDS 1
typedef __attribute__((address_space(1))) v4i* gv4p;   // global v4i*
typedef __attribute__((address_space(3))) v4i* lv4p;   // LDS v4i*
__device__ __forceinline__ void async_b128(const void* g, void* l) {
  __builtin_amdgcn_global_load_async_to_lds_b128(
      (gv4p)(void*)(unsigned long long)(const char*)g,
      (lv4p)l, 0, 0);
}
__device__ __forceinline__ void wait_async0() {
#if __has_builtin(__builtin_amdgcn_s_wait_asynccnt)
  __builtin_amdgcn_s_wait_asynccnt(0);
#else
  asm volatile("s_wait_asynccnt 0" ::: "memory");
#endif
}
#else
#define HAVE_ASYNC_LDS 0
#endif

// ---------------------------------------------------------------------------
// f32 -> bf16 elementwise conversion
// ---------------------------------------------------------------------------
__global__ void cvt_f32_to_bf16_kernel(const float* __restrict__ src,
                                       u16* __restrict__ dst, int n) {
  int i = blockIdx.x * blockDim.x + threadIdx.x;
  if (i < n) dst[i] = f32_to_bf16(src[i]);
}

// ---------------------------------------------------------------------------
// GEMM: Y[m][n] = sum_k A[m][k] * W[n][k] + bias[n]   (A bf16, W bf16)
//   M = 8192, N = K = 1024. Block tile 64x64 (4 waves, wave = 16 rows x 64 cols)
// MODE 0: store bf16 to [B,H,S,Dh]    (Q, K projections)
// MODE 1: store bf16 to [B,H,Dh,S]    (V projection, transposed for P*V)
// MODE 2: store f32  to [M,N]         (output projection -> d_out)
// ---------------------------------------------------------------------------
template <int MODE>
__global__ __launch_bounds__(128)
void proj_gemm_kernel(const u16* __restrict__ A,
                      const u16* __restrict__ W,
                      const float* __restrict__ bias,
                      void* __restrict__ outp) {
  __shared__ u16 As[64][32];
  __shared__ u16 Ws[64][32];

  const int tid  = threadIdx.x;
  const int lane = tid & 31;
  const int wid  = tid >> 5;
  const int half = lane >> 4;
  const int l16  = lane & 15;

  const int m0 = blockIdx.y * 64;
  const int n0 = blockIdx.x * 64;

  const int arow = tid >> 1;         // 0..63
  const int acol = (tid & 1) * 16;   // 0 or 16 (bf16 elements)

  v8f acc[4];
  #pragma unroll
  for (int i = 0; i < 4; ++i) acc[i] = zero8();

  for (int ks = 0; ks < 32; ++ks) {
    const int k0 = ks * 32;
    __syncthreads();
    // Stage A (64x32) and W (64x32) tiles; 32B per thread per tile
#if HAVE_ASYNC_LDS
    async_b128(&A[(m0 + arow) * 1024 + k0 + acol],     &As[arow][acol]);
    async_b128(&A[(m0 + arow) * 1024 + k0 + acol + 8], &As[arow][acol + 8]);
    async_b128(&W[(n0 + arow) * 1024 + k0 + acol],     &Ws[arow][acol]);
    async_b128(&W[(n0 + arow) * 1024 + k0 + acol + 8], &Ws[arow][acol + 8]);
#else
    *(uint4*)&As[arow][acol]     = *(const uint4*)&A[(m0 + arow) * 1024 + k0 + acol];
    *(uint4*)&As[arow][acol + 8] = *(const uint4*)&A[(m0 + arow) * 1024 + k0 + acol + 8];
    *(uint4*)&Ws[arow][acol]     = *(const uint4*)&W[(n0 + arow) * 1024 + k0 + acol];
    *(uint4*)&Ws[arow][acol + 8] = *(const uint4*)&W[(n0 + arow) * 1024 + k0 + acol + 8];
#endif
    if (ks + 1 < 32) {  // hint next K-slab into cache (global_prefetch_b8)
      __builtin_prefetch(&A[(m0 + arow) * 1024 + k0 + 32 + acol], 0, 1);
      __builtin_prefetch(&W[(n0 + arow) * 1024 + k0 + 32 + acol], 0, 1);
    }
#if HAVE_ASYNC_LDS
    wait_async0();
#endif
    __syncthreads();

    // A fragment: 16x32, row = l16 (wave-local), K chunks of 8 per §7.12.2
    Frag16 af;
    const int ml = wid * 16 + l16;
    af.q[0] = *(const uint4*)&As[ml][8 * half];
    af.q[1] = *(const uint4*)&As[ml][16 + 8 * half];

    #pragma unroll
    for (int nt = 0; nt < 4; ++nt) {
      Frag16 bf;
      const int nl = nt * 16 + l16;           // B col N = l16; K chunks of 8 (table order)
      bf.q[0] = *(const uint4*)&Ws[nl][8 * half];
      bf.q[1] = *(const uint4*)&Ws[nl][16 + 8 * half];
      acc[nt] = __builtin_amdgcn_wmma_f32_16x16x32_bf16(
          false, af.bf, false, bf.bf, (short)0, acc[nt], false, false);
    }
  }

  #pragma unroll
  for (int nt = 0; nt < 4; ++nt) {
    const int ng = n0 + nt * 16 + l16;
    const float bv = bias[ng];
    #pragma unroll
    for (int r = 0; r < 8; ++r) {
      const int mg = m0 + wid * 16 + r + 8 * half;   // C/D: M = r + 8*half
      const float val = acc[nt][r] + bv;
      if (MODE == 2) {
        ((float*)outp)[(size_t)mg * 1024 + ng] = val;
      } else {
        const int b = mg >> 11, s = mg & 2047;
        const int h = ng >> 6,  dh = ng & 63;
        u16* o = (u16*)outp;
        if (MODE == 0)
          o[(((size_t)(b * 16 + h) * 2048 + s) * 64) + dh] = f32_to_bf16(val);
        else
          o[(((size_t)(b * 16 + h) * 64 + dh) * 2048) + s] = f32_to_bf16(val);
      }
    }
  }
}

// ---------------------------------------------------------------------------
// Fused attention for one (b, h, 16-row q-tile):
//   scores = Q K^T / 8  (+ causal & key-padding mask) -> LDS (16 x 2048 f32)
//   softmax rows in LDS, write probs (f32) to attn output
//   context = P * V  (P re-read from LDS as bf16), write bf16 ctx [B,S,D]
// Block = 128 threads (4 waves), dynamic LDS = 16*2048*4 = 128 KB.
// ---------------------------------------------------------------------------
__global__ __launch_bounds__(128)
void attention_kernel(const u16* __restrict__ Qp,            // [B,H,S,64] bf16
                      const u16* __restrict__ Kp,            // [B,H,S,64] bf16
                      const u16* __restrict__ Vt,            // [B,H,64,S] bf16
                      const unsigned char* __restrict__ kpm, // [B,S] bool
                      float* __restrict__ attn,              // [B,H,S,S] f32
                      u16* __restrict__ ctx) {               // [B,S,1024] bf16
  extern __shared__ float sc[];  // [16][SEQ]

  const int tid  = threadIdx.x;
  const int lane = tid & 31;
  const int wid  = tid >> 5;
  const int half = lane >> 4;
  const int l16  = lane & 15;

  const int qt = blockIdx.x;   // 0..127
  const int bh = blockIdx.y;   // 0..63
  const int b  = bh >> 4;
  const int h  = bh & 15;
  const int q0 = qt * 16;

  const u16* Qb = Qp + (size_t)bh * SEQ * 64;
  const u16* Kb = Kp + (size_t)bh * SEQ * 64;
  const u16* Vb = Vt + (size_t)bh * 64 * SEQ;

  // Q A-fragments for d in [0,32) and [32,64)
  Frag16 qf[2];
  #pragma unroll
  for (int dblk = 0; dblk < 2; ++dblk) {
    const u16* qrow = Qb + (size_t)(q0 + l16) * 64 + dblk * 32;
    qf[dblk].q[0] = *(const uint4*)(qrow + 8 * half);
    qf[dblk].q[1] = *(const uint4*)(qrow + 16 + 8 * half);
  }

  // ---- Phase 1: 16x16 score tiles, 32 k-tiles per wave ----
  for (int kt = wid; kt < SEQ / 16; kt += 4) {
    const int kcol = kt * 16 + l16;   // B-frag column N = l16
    const u16* krow = Kb + (size_t)kcol * 64;
    Frag16 kf0, kf1;                  // B = K^T: rows d contiguous in memory
    kf0.q[0] = *(const uint4*)(krow + 8 * half);
    kf0.q[1] = *(const uint4*)(krow + 16 + 8 * half);
    kf1.q[0] = *(const uint4*)(krow + 32 + 8 * half);
    kf1.q[1] = *(const uint4*)(krow + 48 + 8 * half);

    v8f c = zero8();
    c = __builtin_amdgcn_wmma_f32_16x16x32_bf16(false, qf[0].bf, false, kf0.bf,
                                                (short)0, c, false, false);
    c = __builtin_amdgcn_wmma_f32_16x16x32_bf16(false, qf[1].bf, false, kf1.bf,
                                                (short)0, c, false, false);

    const bool mk = (kpm[b * SEQ + kcol] != 0);
    #pragma unroll
    for (int r = 0; r < 8; ++r) {
      const int m  = r + 8 * half;
      const int qg = q0 + m;
      float v = c[r] * 0.125f;                 // / sqrt(64)
      if (mk || (kcol > qg)) v = -__builtin_inff();
      sc[m * SEQ + kcol] = v;
    }
  }
  __syncthreads();

  // ---- Phase 2: row softmax in LDS; stream normalized probs to HBM ----
  #pragma unroll
  for (int r4 = 0; r4 < 4; ++r4) {
    const int m = wid * 4 + r4;
    float* row = sc + m * SEQ;
    float mx = -__builtin_inff();
    for (int k = lane; k < SEQ; k += 32) mx = fmaxf(mx, row[k]);
    #pragma unroll
    for (int off = 16; off > 0; off >>= 1) mx = fmaxf(mx, __shfl_xor(mx, off, 32));
    float sum = 0.0f;
    for (int k = lane; k < SEQ; k += 32) {
      float e = __expf(row[k] - mx);
      row[k] = e;
      sum += e;
    }
    #pragma unroll
    for (int off = 16; off > 0; off >>= 1) sum += __shfl_xor(sum, off, 32);
    const float inv = 1.0f / sum;
    float* arow = attn + ((size_t)(bh * SEQ + q0 + m)) * SEQ;
    for (int k = lane; k < SEQ; k += 32) {
      float p = row[k] * inv;
      row[k]  = p;      // keep normalized probs for P*V
      arow[k] = p;      // coalesced f32 output row
    }
  }
  __syncthreads();

  // ---- Phase 3: context = P * V ; wave wid owns d-columns [wid*16, wid*16+16) ----
  v8f cacc = zero8();
  const int d0 = wid * 16;
  for (int kc = 0; kc < SEQ; kc += 32) {
    // A fragment (16x32) of probs from LDS, f32 -> bf16
    const float* prow = sc + l16 * SEQ + kc;
    float4 p0 = *(const float4*)(prow + 8 * half);
    float4 p1 = *(const float4*)(prow + 8 * half + 4);
    float4 p2 = *(const float4*)(prow + 16 + 8 * half);
    float4 p3 = *(const float4*)(prow + 16 + 8 * half + 4);
    float pv[16] = {p0.x, p0.y, p0.z, p0.w, p1.x, p1.y, p1.z, p1.w,
                    p2.x, p2.y, p2.z, p2.w, p3.x, p3.y, p3.z, p3.w};
    Frag16 af;
    #pragma unroll
    for (int e = 0; e < 16; ++e) af.u[e] = f32_to_bf16(pv[e]);

    // B fragment from V^T: col N = l16 -> d, K contiguous along s
    Frag16 vf;
    const u16* vrow = Vb + (size_t)(d0 + l16) * SEQ + kc + 8 * half;
    vf.q[0] = *(const uint4*)vrow;
    vf.q[1] = *(const uint4*)(vrow + 16);

    cacc = __builtin_amdgcn_wmma_f32_16x16x32_bf16(
        false, af.bf, false, vf.bf, (short)0, cacc, false, false);
  }

  #pragma unroll
  for (int r = 0; r < 8; ++r) {
    const int s = q0 + r + 8 * half;
    ctx[(size_t)(b * SEQ + s) * 1024 + h * 64 + d0 + l16] = f32_to_bf16(cacc[r]);
  }
}

// ---------------------------------------------------------------------------
// Host launcher
// ---------------------------------------------------------------------------
extern "C" void kernel_launch(void* const* d_in, const int* in_sizes, int n_in,
                              void* d_out, int out_size, void* d_ws, size_t ws_size,
                              hipStream_t stream) {
  (void)in_sizes; (void)n_in; (void)out_size; (void)ws_size;

  const float*         x   = (const float*)d_in[0];
  const unsigned char* kpm = (const unsigned char*)d_in[1];  // bool mask [B,S]
  const float* W_q = (const float*)d_in[2];
  const float* b_q = (const float*)d_in[3];
  const float* W_k = (const float*)d_in[4];
  const float* b_k = (const float*)d_in[5];
  const float* W_v = (const float*)d_in[6];
  const float* b_v = (const float*)d_in[7];
  const float* W_o = (const float*)d_in[8];
  const float* b_o = (const float*)d_in[9];

  float* out  = (float*)d_out;                                 // [B,S,D]
  float* attn = out + (size_t)BATCH * SEQ * D_MODEL;           // [B,H,S,S]

  // Workspace layout (bf16 elements)
  u16* ws = (u16*)d_ws;
  const size_t XB = (size_t)M_ROWS * D_MODEL;      // 8,388,608
  const size_t WB = (size_t)D_MODEL * D_MODEL;     // 1,048,576
  u16* xb  = ws;
  u16* wqb = xb  + XB;
  u16* wkb = wqb + WB;
  u16* wvb = wkb + WB;
  u16* wob = wvb + WB;
  u16* Qp  = wob + WB;   // [B,H,S,64]
  u16* Kp  = Qp  + XB;   // [B,H,S,64]
  u16* Vt  = Kp  + XB;   // [B,H,64,S]
  u16* ctx = Vt  + XB;   // [B,S,1024]

  // 1) f32 -> bf16 conversions
  cvt_f32_to_bf16_kernel<<<(int)(XB / 256), 256, 0, stream>>>(x,   xb,  (int)XB);
  cvt_f32_to_bf16_kernel<<<(int)(WB / 256), 256, 0, stream>>>(W_q, wqb, (int)WB);
  cvt_f32_to_bf16_kernel<<<(int)(WB / 256), 256, 0, stream>>>(W_k, wkb, (int)WB);
  cvt_f32_to_bf16_kernel<<<(int)(WB / 256), 256, 0, stream>>>(W_v, wvb, (int)WB);
  cvt_f32_to_bf16_kernel<<<(int)(WB / 256), 256, 0, stream>>>(W_o, wob, (int)WB);

  // 2) QKV projections (WMMA)
  dim3 pg(D_MODEL / 64, M_ROWS / 64);  // (16, 128)
  proj_gemm_kernel<0><<<pg, 128, 0, stream>>>(xb, wqb, b_q, (void*)Qp);
  proj_gemm_kernel<0><<<pg, 128, 0, stream>>>(xb, wkb, b_k, (void*)Kp);
  proj_gemm_kernel<1><<<pg, 128, 0, stream>>>(xb, wvb, b_v, (void*)Vt);

  // 3) Fused attention (scores + softmax + attn_weights output + P*V)
  dim3 ag(SEQ / 16, BATCH * NUM_HEADS);  // (128, 64)
  attention_kernel<<<ag, 128, 16 * SEQ * sizeof(float), stream>>>(
      Qp, Kp, Vt, kpm, attn, ctx);

  // 4) Output projection -> f32 d_out
  proj_gemm_kernel<2><<<pg, 128, 0, stream>>>(ctx, wob, b_o, (void*)out);
}
